// DPRNN_64398739636859
// MI455X (gfx1250) — matrix operand order repaired and checked by reading, not compiled, not run on hardware
//
#include <hip/hip_runtime.h>
#include <hip/hip_bf16.h>

// Seq2seq LSTM (B=8192, T=64, H=128, 50 decoder steps) for MI455X / gfx1250.
// Batch-parallel, time-sequential: each workgroup owns 16 batch rows, keeps
// h/c + f16 weights resident in LDS, and runs the recurrent 16x512x128 GEMM
// on v_wmma_f32_16x16x32_f16 (32 WMMAs per wave per step, 4 waves).

#define BB      8192
#define TT      64
#define DIN     8
#define HH      128
#define G4      512      // 4*H gates
#define NSTEPS  50
#define DOUT    8
#define P_DROP  0.2f
#define KEEP    1.25f    // 1/(1-0.2)

typedef __attribute__((ext_vector_type(16))) _Float16 v16h;
typedef __attribute__((ext_vector_type(8)))  _Float16 v8h;
typedef __attribute__((ext_vector_type(8)))  float    v8f;

// ---- dynamic LDS layout (bytes) ----
#define WHH16_OFF 0                         // 512*128 f16      = 131072
#define WIH_OFF   (WHH16_OFF + 131072)      // 512*8   f32      =  16384
#define BIAS_OFF  (WIH_OFF   + 16384)       // 512     f32      =   2048
#define OUTW_OFF  (BIAS_OFF  + 2048)        // 8*128   f32      =   4096
#define OUTB_OFF  (OUTW_OFF  + 4096)        // 8 f32 (pad 64)   =     64
#define GATES_OFF (OUTB_OFF  + 64)          // 16*512  f32      =  32768
#define H16_OFF   (GATES_OFF + 32768)       // 16*128  f16      =   4096
#define H32_OFF   (H16_OFF   + 4096)        // 16*128  f32      =   8192
#define C32_OFF   (H32_OFF   + 8192)        // 16*128  f32      =   8192
#define HD_OFF    (C32_OFF   + 8192)        // 16*128  f32      =   8192
#define XIN_OFF   (HD_OFF    + 8192)        // 16*8    f32      =    512
#define SMEM_TOTAL (XIN_OFF  + 512)         // 215616 B (< 320 KB/WGP)

static __device__ __forceinline__ float sigm(float v) {
  return 1.0f / (1.0f + __expf(-v));
}

static __device__ __forceinline__ v8f wmma_f16(v16h a, v16h b, v8f c) {
  return __builtin_amdgcn_wmma_f32_16x16x32_f16(
      /*neg_a=*/false, a, /*neg_b=*/false, b,
      /*c_mod=*/(short)0, c, /*reuse_a=*/false, /*reuse_b=*/false);
}

// gates[16x512] = h16[16x128] @ Whh16^T + bias.  One wave covers 8 N-tiles.
static __device__ __forceinline__ void gemm_gates(
    const _Float16* __restrict__ whh16, const float* __restrict__ bias,
    const _Float16* __restrict__ h16, float* __restrict__ gates,
    int lane, int wv) {
  const int mrow   = lane & 15;
  const int hiHalf = lane >> 4;           // 0: lanes 0-15, 1: lanes 16-31

  // A fragments (shared across all N-tiles): lane holds row mrow,
  // K = {kb*32 + hiHalf*8 + 0..7, +16..23}  (ISA 16-bit A layout)
  v16h afrag[4];
#pragma unroll
  for (int kb = 0; kb < 4; kb++) {
    const _Float16* abase = &h16[mrow * HH + kb * 32 + hiHalf * 8];
    v8h lo = *(const v8h*)(abase);
    v8h hi = *(const v8h*)(abase + 16);
    v16h a;
#pragma unroll
    for (int i = 0; i < 8; i++) { a[i] = lo[i]; a[i + 8] = hi[i]; }
    afrag[kb] = a;
  }

#pragma unroll
  for (int nt = 0; nt < 8; nt++) {
    const int ntg  = wv * 8 + nt;
    const int ncol = ntg * 16 + mrow;     // this lane's output column
    const float bv = bias[ncol];
    v8f acc;
#pragma unroll
    for (int j = 0; j < 8; j++) acc[j] = bv;
#pragma unroll
    for (int kb = 0; kb < 4; kb++) {
      // B fragment: lane holds column ncol,
      // K = {kb*32 + hiHalf*8 + 0..7, +16..23} (matches A layout).
      const _Float16* bbase = &whh16[ncol * HH + kb * 32 + hiHalf * 8];
      v8h lo = *(const v8h*)(bbase);
      v8h hi = *(const v8h*)(bbase + 16);
      v16h b;
#pragma unroll
      for (int i = 0; i < 8; i++) { b[i] = lo[i]; b[i + 8] = hi[i]; }
      acc = wmma_f16(afrag[kb], b, acc);
    }
    // C/D layout: VGPR j <-> row (j + hiHalf*8), col ncol
#pragma unroll
    for (int j = 0; j < 8; j++)
      gates[(j + hiHalf * 8) * G4 + ncol] = acc[j];
  }
}

// Elementwise LSTM cell: adds the tiny x@Wih^T (K=8) term, applies gate
// nonlinearities, updates c/h.  Thread tid covers m=tid/8, n=(tid%8)*16+r.
// If hd!=nullptr also produces the dropout-masked hidden (decoder path).
static __device__ __forceinline__ void cell_update(
    float* __restrict__ gates, const float* __restrict__ wih,
    const float* __restrict__ xin, float* __restrict__ c32,
    float* __restrict__ h32, _Float16* __restrict__ h16,
    float* __restrict__ hd, const float* u, int tid) {
  const int m  = tid >> 3;
  const int n0 = (tid & 7) * 16;
  float xr[DIN];
#pragma unroll
  for (int d = 0; d < DIN; d++) xr[d] = xin[m * DIN + d];
#pragma unroll
  for (int r = 0; r < 16; r++) {
    const int n = n0 + r;
    float gi = gates[m * G4 + n];
    float gf = gates[m * G4 + 128 + n];
    float gg = gates[m * G4 + 256 + n];
    float go = gates[m * G4 + 384 + n];
#pragma unroll
    for (int d = 0; d < DIN; d++) {
      const float xv = xr[d];
      gi += xv * wih[n * DIN + d];
      gf += xv * wih[(n + 128) * DIN + d];
      gg += xv * wih[(n + 256) * DIN + d];
      go += xv * wih[(n + 384) * DIN + d];
    }
    const float c = sigm(gf) * c32[m * HH + n] + sigm(gi) * tanhf(gg);
    const float h = sigm(go) * tanhf(c);
    c32[m * HH + n] = c;
    h32[m * HH + n] = h;
    h16[m * HH + n] = (_Float16)h;
    if (hd) hd[m * HH + n] = h * (u[r] >= P_DROP ? KEEP : 0.0f);
  }
}

__global__ __launch_bounds__(128) void dprnn_seq2seq_kernel(
    const float* __restrict__ x, const float* __restrict__ drop_u,
    const float* __restrict__ enc_Wih, const float* __restrict__ enc_Whh,
    const float* __restrict__ enc_b, const float* __restrict__ dec_Wih,
    const float* __restrict__ dec_Whh, const float* __restrict__ dec_b,
    const float* __restrict__ out_W, const float* __restrict__ out_b,
    float* __restrict__ out) {
  extern __shared__ char smem[];
  _Float16* whh16 = (_Float16*)(smem + WHH16_OFF);
  float*    wih   = (float*)(smem + WIH_OFF);
  float*    bias  = (float*)(smem + BIAS_OFF);
  float*    outW  = (float*)(smem + OUTW_OFF);
  float*    outb  = (float*)(smem + OUTB_OFF);
  float*    gates = (float*)(smem + GATES_OFF);
  _Float16* h16   = (_Float16*)(smem + H16_OFF);
  float*    h32   = (float*)(smem + H32_OFF);
  float*    c32   = (float*)(smem + C32_OFF);
  float*    hd    = (float*)(smem + HD_OFF);
  float*    xin   = (float*)(smem + XIN_OFF);

  const int tid  = threadIdx.x;
  const int lane = tid & 31;
  const int wv   = tid >> 5;
  const int b0   = blockIdx.x * 16;

  // ---- init: encoder weights -> LDS (f16 image of Whh, f32 rest) ----
  for (int i = tid; i < G4 * HH; i += 128) whh16[i] = (_Float16)enc_Whh[i];
  for (int i = tid; i < G4 * DIN; i += 128) wih[i] = enc_Wih[i];
  for (int i = tid; i < G4; i += 128) bias[i] = enc_b[i];
  for (int i = tid; i < DOUT * HH; i += 128) outW[i] = out_W[i];
  if (tid < DOUT) outb[tid] = out_b[tid];
  for (int i = tid; i < 16 * HH; i += 128) {
    c32[i] = 0.0f; h32[i] = 0.0f; h16[i] = (_Float16)0.0f;
  }
  __syncthreads();

  // ---- encoder: 64 sequential steps ----
  for (int t = 0; t < TT; t++) {
    { // stage x_t (overlaps with WMMA phase; consumed after the barrier)
      const int m = tid >> 3, d = tid & 7;
      xin[m * DIN + d] = x[(size_t)(b0 + m) * TT * DIN + t * DIN + d];
    }
    gemm_gates(whh16, bias, h16, gates, lane, wv);
    __syncthreads();
    cell_update(gates, wih, xin, c32, h32, h16, nullptr, nullptr, tid);
    __syncthreads();
  }

  // ---- swap to decoder weights; seed decoder input ----
  for (int i = tid; i < G4 * HH; i += 128) whh16[i] = (_Float16)dec_Whh[i];
  for (int i = tid; i < G4 * DIN; i += 128) wih[i] = dec_Wih[i];
  for (int i = tid; i < G4; i += 128) bias[i] = dec_b[i];
  {
    const int m = tid >> 3, d = tid & 7;
    xin[m * DIN + d] = x[(size_t)(b0 + m) * TT * DIN + (TT - 1) * DIN + d];
  }
  __syncthreads();

  // ---- decoder: 50 sequential steps ----
  for (int s = 0; s < NSTEPS; s++) {
    // Issue this step's dropout-mask loads during the WMMA phase and
    // prefetch next step's slab (global_prefetch_b8).
    float u[16];
    {
      const int m = tid >> 3, k0 = (tid & 7) * 16;
      const float* up = &drop_u[((size_t)s * BB + b0 + m) * HH + k0];
#pragma unroll
      for (int j = 0; j < 16; j++) u[j] = up[j];
      if (s + 1 < NSTEPS)
        __builtin_prefetch(&drop_u[((size_t)(s + 1) * BB + b0 + m) * HH + k0],
                           0, 1);
    }
    gemm_gates(whh16, bias, h16, gates, lane, wv);
    __syncthreads();
    cell_update(gates, wih, xin, c32, h32, h16, hd, u, tid);
    __syncthreads();
    { // projection y = hd @ out_W^T + out_b ; y feeds back as next input
      const int m = tid >> 3, d = tid & 7;
      float acc = outb[d];
#pragma unroll 8
      for (int k = 0; k < HH; k++) acc += hd[m * HH + k] * outW[d * HH + k];
      out[(size_t)(b0 + m) * NSTEPS * DOUT + s * DOUT + d] = acc;
      xin[m * DIN + d] = acc;
      // No barrier needed here: next phase's gates writes are disjoint from
      // hd reads, and xin is consumed only after the next post-GEMM barrier.
    }
  }
}

extern "C" void kernel_launch(void* const* d_in, const int* in_sizes, int n_in,
                              void* d_out, int out_size, void* d_ws,
                              size_t ws_size, hipStream_t stream) {
  const float* x       = (const float*)d_in[0];
  const float* drop_u  = (const float*)d_in[1];
  const float* enc_Wih = (const float*)d_in[2];
  const float* enc_Whh = (const float*)d_in[3];
  const float* enc_b   = (const float*)d_in[4];
  const float* dec_Wih = (const float*)d_in[5];
  const float* dec_Whh = (const float*)d_in[6];
  const float* dec_b   = (const float*)d_in[7];
  const float* out_W   = (const float*)d_in[8];
  const float* out_b   = (const float*)d_in[9];
  float* out = (float*)d_out;

  const int nB   = in_sizes[0] / (TT * DIN);  // 8192
  const int grid = nB / 16;                   // 512 workgroups
  dprnn_seq2seq_kernel<<<grid, 128, SMEM_TOTAL, stream>>>(
      x, drop_u, enc_Wih, enc_Whh, enc_b, dec_Wih, dec_Whh, dec_b, out_W,
      out_b, out);
}
